// SAGEConvMLPModel_21981642620997
// MI455X (gfx1250) — matrix operand order repaired and checked by reading, not compiled, not run on hardware
//
#include <hip/hip_runtime.h>

typedef __attribute__((ext_vector_type(16))) __bf16 v16bf;
typedef __attribute__((ext_vector_type(8)))  float  v8f;

#define N_NODES 50000
#define N_EDGES 400000
#define IN_CH   1024
#define FEAT_CH 128
#define HID     32
#define OUT_CH  3
#define BN_EPS  1e-5f
#define SLOPE   0.01f

// ---------------------------------------------------------------------------
// Zero a float4 region (agg + cnt must be zeroed every call).
// ---------------------------------------------------------------------------
__global__ __launch_bounds__(256) void zero_f4(float4* __restrict__ p, int n4) {
    int i = blockIdx.x * 256 + threadIdx.x;
    if (i < n4) p[i] = make_float4(0.f, 0.f, 0.f, 0.f);
}

// ---------------------------------------------------------------------------
// Convert W_l / W_r (fp32 row-major [128 x 1024]) to bf16, same layout.
// ---------------------------------------------------------------------------
__global__ __launch_bounds__(256) void convert_weights(
    const float* __restrict__ wl_f, const float* __restrict__ wr_f,
    __bf16* __restrict__ wl, __bf16* __restrict__ wr) {
    int i = blockIdx.x * 256 + threadIdx.x;
    if (i < FEAT_CH * IN_CH) {
        wl[i] = (__bf16)wl_f[i];
        wr[i] = (__bf16)wr_f[i];
    }
}

// ---------------------------------------------------------------------------
// Dual GEMM with WMMA bf16: Yl = feat @ W_l.T, Yr = feat @ W_r.T
// Block = 256 threads = 8 waves. Block owns a 16-row node tile; wave w owns
// output columns [16w, 16w+16) of BOTH Yl and Yr (A operand reused).
// A layout: lane holds row lane&15, 16 contiguous K at 16*hi (same order as B).
// B layout: lane holds col lane&15, 16 contiguous K at 16*hi.
// ---------------------------------------------------------------------------
__global__ __launch_bounds__(256) void gemm_dual_wmma(
    const float*  __restrict__ feat,
    const __bf16* __restrict__ wl,
    const __bf16* __restrict__ wr,
    float* __restrict__ yl, float* __restrict__ yr) {
    const int lane = threadIdx.x & 31;
    const int wave = threadIdx.x >> 5;          // 0..7
    const int hi   = lane >> 4;                 // 0/1
    const int l15  = lane & 15;
    const int tile = blockIdx.x;                // 0..3124 (3125*16 == 50000)
    const int rowA = tile * 16 + l15;           // node row this lane feeds into A
    const int col  = wave * 16 + l15;           // output column (weight row)

    const float*  arow  = feat + (size_t)rowA * IN_CH + 16 * hi;
    const __bf16* blrow = wl   + (size_t)col  * IN_CH + 16 * hi;
    const __bf16* brrow = wr   + (size_t)col  * IN_CH + 16 * hi;

    v8f accL = {};
    v8f accR = {};

    #pragma unroll 4
    for (int kb = 0; kb < IN_CH; kb += 32) {
        // A: 16 contiguous floats (K = kb+16hi+0..15)
        float4 f0 = *(const float4*)(arow + kb);
        float4 f1 = *(const float4*)(arow + kb + 4);
        float4 f2 = *(const float4*)(arow + kb + 8);
        float4 f3 = *(const float4*)(arow + kb + 12);
        v16bf a;
        a[0]  = (__bf16)f0.x; a[1]  = (__bf16)f0.y; a[2]  = (__bf16)f0.z; a[3]  = (__bf16)f0.w;
        a[4]  = (__bf16)f1.x; a[5]  = (__bf16)f1.y; a[6]  = (__bf16)f1.z; a[7]  = (__bf16)f1.w;
        a[8]  = (__bf16)f2.x; a[9]  = (__bf16)f2.y; a[10] = (__bf16)f2.z; a[11] = (__bf16)f2.w;
        a[12] = (__bf16)f3.x; a[13] = (__bf16)f3.y; a[14] = (__bf16)f3.z; a[15] = (__bf16)f3.w;

        // B: 16 contiguous bf16 (32B aligned) per lane, per weight matrix
        v16bf bl = *(const v16bf*)(blrow + kb);
        v16bf br = *(const v16bf*)(brrow + kb);

        accL = __builtin_amdgcn_wmma_f32_16x16x32_bf16(
                   false, a, false, bl, (short)0, accL, false, false);
        accR = __builtin_amdgcn_wmma_f32_16x16x32_bf16(
                   false, a, false, br, (short)0, accR, false, false);
    }

    // D layout: acc[r] -> row (r + 8*hi), col (lane&15)
    #pragma unroll
    for (int r = 0; r < 8; ++r) {
        int m = tile * 16 + r + 8 * hi;
        yl[(size_t)m * FEAT_CH + col] = accL[r];
        yr[(size_t)m * FEAT_CH + col] = accR[r];
    }
}

// ---------------------------------------------------------------------------
// Per-target-node in-degree (as float, for the mean).
// ---------------------------------------------------------------------------
__global__ __launch_bounds__(256) void count_edges(
    const int* __restrict__ dst, float* __restrict__ cnt) {
    int e = blockIdx.x * 256 + threadIdx.x;
    if (e < N_EDGES) atomicAdd(&cnt[dst[e]], 1.0f);
}

// ---------------------------------------------------------------------------
// Scatter-add Yl[src] rows (128 f32) into agg[dst]. One thread = 4 channels.
// ---------------------------------------------------------------------------
__global__ __launch_bounds__(256) void scatter_add(
    const int* __restrict__ src, const int* __restrict__ dst,
    const float* __restrict__ yl, float* __restrict__ agg) {
    int idx = blockIdx.x * 256 + threadIdx.x;   // N_EDGES * 32 threads
    int e = idx >> 5;
    int c = (idx & 31) * 4;
    if (e < N_EDGES) {
        int s = src[e], d = dst[e];
        float4 v = *(const float4*)(yl + (size_t)s * FEAT_CH + c);
        float* ap = agg + (size_t)d * FEAT_CH + c;
        atomicAdd(ap + 0, v.x);
        atomicAdd(ap + 1, v.y);
        atomicAdd(ap + 2, v.z);
        atomicAdd(ap + 3, v.w);
    }
}

// ---------------------------------------------------------------------------
// Fused tail: x = agg/max(cnt,1) + b_l + Yr ; leaky ; fc1 ; relu ; BN ; fc2.
// Block = 128 threads = 4 waves; wave handles one node. fc1_w staged in LDS
// transposed ([c][j]) so lane-j reads are bank-conflict-free.
// ---------------------------------------------------------------------------
__global__ __launch_bounds__(128) void final_mlp(
    const float* __restrict__ agg,  const float* __restrict__ cnt,
    const float* __restrict__ yr,   const float* __restrict__ b_l,
    const float* __restrict__ fc1_w, const float* __restrict__ fc1_b,
    const float* __restrict__ fc2_w, const float* __restrict__ fc2_b,
    const float* __restrict__ g,    const float* __restrict__ be,
    const float* __restrict__ mu,   const float* __restrict__ var,
    float* __restrict__ out) {
    __shared__ float sW1t[FEAT_CH * HID];   // [c][j]
    __shared__ float sW2[OUT_CH * HID];
    __shared__ float sbl[FEAT_CH];
    __shared__ float sb1[HID], sg[HID], sbe[HID], smu[HID], svar[HID];
    __shared__ float sb2[OUT_CH];
    __shared__ float xs[4][FEAT_CH];
    __shared__ float hs[4][HID];

    int tid = threadIdx.x;
    for (int i = tid; i < HID * FEAT_CH; i += 128) {
        int j = i >> 7, c = i & 127;        // fc1_w row-major [32 x 128]
        sW1t[c * HID + j] = fc1_w[i];
    }
    if (tid < FEAT_CH) sbl[tid] = b_l[tid];
    if (tid < OUT_CH * HID) sW2[tid] = fc2_w[tid];
    if (tid < HID) {
        sb1[tid] = fc1_b[tid]; sg[tid]  = g[tid];  sbe[tid] = be[tid];
        smu[tid] = mu[tid];    svar[tid] = var[tid];
    }
    if (tid < OUT_CH) sb2[tid] = fc2_b[tid];
    __syncthreads();

    int wave = tid >> 5, lane = tid & 31;
    int node = blockIdx.x * 4 + wave;       // 12500 * 4 == 50000
    int c = lane * 4;

    float inv = 1.0f / fmaxf(cnt[node], 1.0f);
    float4 a = *(const float4*)(agg + (size_t)node * FEAT_CH + c);
    float4 y = *(const float4*)(yr  + (size_t)node * FEAT_CH + c);
    float xv[4] = { a.x * inv + sbl[c + 0] + y.x,
                    a.y * inv + sbl[c + 1] + y.y,
                    a.z * inv + sbl[c + 2] + y.z,
                    a.w * inv + sbl[c + 3] + y.w };
    #pragma unroll
    for (int i = 0; i < 4; ++i) {
        float v = xv[i];
        xs[wave][c + i] = (v > 0.0f) ? v : SLOPE * v;
    }
    __syncthreads();

    // fc1: lane j computes h[j]
    float s = sb1[lane];
    #pragma unroll 8
    for (int cc = 0; cc < FEAT_CH; ++cc)
        s += sW1t[cc * HID + lane] * xs[wave][cc];
    s = fmaxf(s, 0.0f);
    s = (s - smu[lane]) * rsqrtf(svar[lane] + BN_EPS) * sg[lane] + sbe[lane];
    hs[wave][lane] = s;
    __syncthreads();

    if (lane < OUT_CH) {
        float o = sb2[lane];
        #pragma unroll
        for (int j = 0; j < HID; ++j) o += sW2[lane * HID + j] * hs[wave][j];
        out[(size_t)node * OUT_CH + lane] = o;
    }
}

// ---------------------------------------------------------------------------
extern "C" void kernel_launch(void* const* d_in, const int* in_sizes, int n_in,
                              void* d_out, int out_size, void* d_ws, size_t ws_size,
                              hipStream_t stream) {
    const float* feat   = (const float*)d_in[0];
    const int*   edges2 = (const int*)  d_in[2];   // [2, N_EDGES], int32
    const float* W_l    = (const float*)d_in[5];
    const float* b_l    = (const float*)d_in[6];
    const float* W_r    = (const float*)d_in[7];
    const float* fc1_w  = (const float*)d_in[8];
    const float* fc1_b  = (const float*)d_in[9];
    const float* fc2_w  = (const float*)d_in[10];
    const float* fc2_b  = (const float*)d_in[11];
    const float* bn_g   = (const float*)d_in[12];
    const float* bn_b   = (const float*)d_in[13];
    const float* bn_m   = (const float*)d_in[14];
    const float* bn_v   = (const float*)d_in[15];
    float* out = (float*)d_out;

    // Workspace layout (all 256B-aligned offsets):
    //   [0)        Wl bf16: 128*1024*2 = 262144 B
    //   [262144)   Wr bf16: 262144 B
    //   [524288)   Yl f32 : 50000*128*4 = 25,600,000 B
    //   [+Yl)      Yr f32 : 25,600,000 B
    //   [+Yr)      agg f32: 25,600,000 B
    //   [+agg)     cnt f32: 200,000 B          (total ~77.5 MB)
    char* ws = (char*)d_ws;
    const size_t YB = (size_t)N_NODES * FEAT_CH * sizeof(float);
    __bf16* wl_bf = (__bf16*)ws;
    __bf16* wr_bf = (__bf16*)(ws + 262144);
    float*  yl    = (float*)(ws + 524288);
    float*  yr    = (float*)(ws + 524288 + YB);
    float*  agg   = (float*)(ws + 524288 + 2 * YB);
    float*  cntp  = (float*)(ws + 524288 + 3 * YB);

    const int* src = edges2;
    const int* dst = edges2 + N_EDGES;

    // agg (6.4M f32) + cnt (50000 f32) are contiguous: zero as one region.
    const int n4 = (N_NODES * FEAT_CH + N_NODES) / 4;   // 1,612,500 float4
    zero_f4<<<(n4 + 255) / 256, 256, 0, stream>>>((float4*)agg, n4);

    convert_weights<<<(FEAT_CH * IN_CH + 255) / 256, 256, 0, stream>>>(
        W_l, W_r, wl_bf, wr_bf);

    gemm_dual_wmma<<<N_NODES / 16, 256, 0, stream>>>(feat, wl_bf, wr_bf, yl, yr);

    count_edges<<<(N_EDGES + 255) / 256, 256, 0, stream>>>(dst, cntp);

    scatter_add<<<(N_EDGES * 32) / 256, 256, 0, stream>>>(src, dst, yl, agg);

    final_mlp<<<N_NODES / 4, 128, 0, stream>>>(
        agg, cntp, yr, b_l, fc1_w, fc1_b, fc2_w, fc2_b,
        bn_g, bn_b, bn_m, bn_v, out);
}
